// MyMultiHeadAttention_83030307766586
// MI455X (gfx1250) — matrix operand order repaired and checked by reading, not compiled, not run on hardware
//
#include <hip/hip_runtime.h>
#include <stdint.h>

// ---------- types ----------
typedef __bf16 bf16;
typedef __attribute__((ext_vector_type(8)))  bf16  v8bf;
typedef __attribute__((ext_vector_type(16))) bf16  v16bf;
typedef __attribute__((ext_vector_type(8)))  float v8f;

#define NB    4
#define TSEQ  2048
#define UDIM  1024
#define NH    16
#define HD    64           // head dim = 1024/16
#define MROWS (NB * TSEQ)  // 8192

// ---------- WMMA fragment loaders (per CDNA5 ISA 7.12.2 layouts) ----------
// A 16x32 bf16: lane m=lane&15, half=lane>>4; elems 0..7 = K=half*8+0..7,
// elems 8..15 = K=16+half*8+0..7 (row-major source, stride ld).
__device__ __forceinline__ v16bf load_a16(const bf16* __restrict__ src,
                                          int row0, int k0, int ld, int lane) {
  int m = lane & 15, h = lane >> 4;
  const bf16* p = src + (size_t)(row0 + m) * ld + k0 + h * 8;
  v8bf lo = *(const v8bf*)p;
  v8bf hi = *(const v8bf*)(p + 16);
  v16bf a;
#pragma unroll
  for (int i = 0; i < 8; ++i) { a[i] = lo[i]; a[i + 8] = hi[i]; }
  return a;
}

// B 32x16 bf16: lane n=lane&15 picks column; half=lane>>4; elems 0..7 =
// K=half*8+0..7, elems 8..15 = K=16+half*8+0..7. Source is B^T row-major [N][K].
__device__ __forceinline__ v16bf load_b16(const bf16* __restrict__ srcT,
                                          int col0, int k0, int ld, int lane) {
  int n = lane & 15, h = lane >> 4;
  const bf16* p = srcT + (size_t)(col0 + n) * ld + k0 + h * 8;
  v8bf lo = *(const v8bf*)p;
  v8bf hi = *(const v8bf*)(p + 16);
  v16bf b;
#pragma unroll
  for (int i = 0; i < 8; ++i) { b[i] = lo[i]; b[i + 8] = hi[i]; }
  return b;
}

__device__ __forceinline__ v8f wmma_bf16(v16bf a, v16bf b, v8f c) {
  return __builtin_amdgcn_wmma_f32_16x16x32_bf16(false, a, false, b,
                                                 (short)0, c, false, false);
}

// ---------- stage 1: casts ----------
__global__ void cast_f32_bf16(const float* __restrict__ x, bf16* __restrict__ y, int n) {
  int i = blockIdx.x * 256 + threadIdx.x;
  if (i < n) y[i] = (bf16)x[i];
}

// Wt[n][k] = W[k][n], cast to bf16. idx = n*K + k (coalesced writes).
__global__ void transpose_cast(const float* __restrict__ W, bf16* __restrict__ Wt,
                               int K, int N) {
  int idx = blockIdx.x * 256 + threadIdx.x;
  if (idx >= K * N) return;
  int n = idx / K, k = idx % K;
  Wt[idx] = (bf16)W[(size_t)k * N + n];
}

// ---------- stage 2/4: bf16 WMMA GEMM ----------
// C[M,N] = alpha * A[M,K] @ B[K,N]; B passed transposed row-major [N][K].
// MODE 0: bf16 store; MODE 1: bf16 transposed store; MODE 2: f32 + bias.
// Block = 256 (8 waves); WG tile 256x64, wave tile 32x64
// (2 A-frags x 4 B-frags -> 8 WMMAs per K-step).
template <int MODE>
__global__ void gemm_bf16_k(const bf16* __restrict__ A, const bf16* __restrict__ Bt,
                            void* __restrict__ Cv, const float* __restrict__ bias,
                            int Kk, float alpha, int ldc) {
  const int lane = threadIdx.x & 31;
  const int w = threadIdx.x >> 5;
  const int row0 = blockIdx.y * 256 + w * 32;
  const int col0 = blockIdx.x * 64;
  v8f acc[2][4] = {};
  for (int k0 = 0; k0 < Kk; k0 += 32) {
    v16bf a0 = load_a16(A, row0, k0, Kk, lane);
    v16bf a1 = load_a16(A, row0 + 16, k0, Kk, lane);
#pragma unroll
    for (int j = 0; j < 4; ++j) {
      v16bf b = load_b16(Bt, col0 + j * 16, k0, Kk, lane);
      acc[0][j] = wmma_bf16(a0, b, acc[0][j]);
      acc[1][j] = wmma_bf16(a1, b, acc[1][j]);
    }
  }
  const int nn = lane & 15, h = lane >> 4;
#pragma unroll
  for (int i2 = 0; i2 < 2; ++i2) {
#pragma unroll
    for (int j = 0; j < 4; ++j) {
      int col = col0 + j * 16 + nn;
#pragma unroll
      for (int r = 0; r < 8; ++r) {
        int row = row0 + i2 * 16 + r + h * 8;
        float v = alpha * acc[i2][j][r];
        if constexpr (MODE == 0) {
          ((bf16*)Cv)[(size_t)row * ldc + col] = (bf16)v;
        } else if constexpr (MODE == 1) {
          ((bf16*)Cv)[(size_t)col * ldc + row] = (bf16)v;
        } else {
          ((float*)Cv)[(size_t)row * ldc + col] = v + bias[col];
        }
      }
    }
  }
}

// ---------- stage 3: fused attention per (batch, 16-query tile) ----------
// Loops over all 16 heads so the head-average of the probabilities can be
// accumulated in LDS (no global atomics). Dynamic LDS: scores 128KB +
// avg 128KB + 4KB phase-3 partials = 260KB (within 320KB/WG on CDNA5).
__global__ void attn_k(const bf16* __restrict__ Qp, const bf16* __restrict__ Kp,
                       const bf16* __restrict__ Vt, const uint8_t* __restrict__ mask,
                       float* __restrict__ avg, bf16* __restrict__ ctx) {
  extern __shared__ float S[];          // [16][TSEQ] scores -> probabilities
  float* AVG = S + 16 * TSEQ;           // [16][TSEQ] head-average accumulator
  float* SC  = AVG + 16 * TSEQ;         // [4][16][16] phase-3 partial tiles
  const int lane = threadIdx.x & 31;
  const int w = threadIdx.x >> 5;       // 0..7
  const int n = blockIdx.y;             // batch
  const int q0 = blockIdx.x * 16;
  const int rowQ = n * TSEQ + q0;
  const int nn = lane & 15, hf = lane >> 4;
  const int row = threadIdx.x >> 4;     // softmax mapping: 16 threads/row
  const int sub = threadIdx.x & 15;

  for (int h = 0; h < NH; ++h) {
    const int kbase = h * HD;

    // Phase 1: S = Q_tile(16x64) @ K^T (64x2048); scale folded into Qp.
    v16bf a0 = load_a16(Qp, rowQ, kbase, UDIM, lane);
    v16bf a1 = load_a16(Qp, rowQ, kbase + 32, UDIM, lane);
    for (int tt = 0; tt < 16; ++tt) {
      int t0 = w * 256 + tt * 16;
      v16bf b0 = load_b16(Kp, n * TSEQ + t0, kbase, UDIM, lane);
      v16bf b1 = load_b16(Kp, n * TSEQ + t0, kbase + 32, UDIM, lane);
      v8f c = {};
      c = wmma_bf16(a0, b0, c);
      c = wmma_bf16(a1, b1, c);
#pragma unroll
      for (int r = 0; r < 8; ++r) S[(r + hf * 8) * TSEQ + t0 + nn] = c[r];
    }
    __syncthreads();

    // Phase 2: row softmax; accumulate p/NH into AVG; p written back to S.
    {
      float m = -__builtin_inff();
      for (int t = sub; t < TSEQ; t += 16) {
        float s = S[row * TSEQ + t];
        if (mask[n * TSEQ + t]) { s = -__builtin_inff(); S[row * TSEQ + t] = s; }
        m = fmaxf(m, s);
      }
#pragma unroll
      for (int off = 8; off; off >>= 1) m = fmaxf(m, __shfl_xor(m, off, 16));
      float sum = 0.f;
      for (int t = sub; t < TSEQ; t += 16) {
        float e = __expf(S[row * TSEQ + t] - m);
        S[row * TSEQ + t] = e;
        sum += e;
      }
#pragma unroll
      for (int off = 8; off; off >>= 1) sum += __shfl_xor(sum, off, 16);
      const float inv = 1.0f / sum;
      if (h == 0) {
        for (int t = sub; t < TSEQ; t += 16) {
          float p = S[row * TSEQ + t] * inv;
          S[row * TSEQ + t] = p;
          AVG[row * TSEQ + t] = p * (1.0f / NH);
        }
      } else {
        for (int t = sub; t < TSEQ; t += 16) {
          float p = S[row * TSEQ + t] * inv;
          S[row * TSEQ + t] = p;
          AVG[row * TSEQ + t] += p * (1.0f / NH);
        }
      }
    }
    __syncthreads();

    // Phase 3: ctx_tile(16x64) = P(16x2048) @ V(2048x64). All 8 waves:
    // wave (dt, kh) = (w&3, w>>2) handles d-tile dt, K-half kh.
    {
      const int dt = w & 3, kh = w >> 2;
      v8f acc = {};
      for (int kk = kh * 1024; kk < kh * 1024 + 1024; kk += 32) {
        const float* p = S + (lane & 15) * TSEQ + kk + hf * 8;
        v16bf a;
#pragma unroll
        for (int i = 0; i < 8; ++i) { a[i] = (bf16)p[i]; a[i + 8] = (bf16)p[i + 16]; }
        v16bf b = load_b16(Vt, kbase + dt * 16, n * TSEQ + kk, MROWS, lane);
        acc = wmma_bf16(a, b, acc);
      }
      if (kh == 1) {
#pragma unroll
        for (int r = 0; r < 8; ++r) SC[dt * 256 + (r + hf * 8) * 16 + nn] = acc[r];
      }
      __syncthreads();
      if (kh == 0) {
#pragma unroll
        for (int r = 0; r < 8; ++r) {
          float v = acc[r] + SC[dt * 256 + (r + hf * 8) * 16 + nn];
          int rowg = q0 + r + hf * 8;
          ctx[((size_t)n * TSEQ + rowg) * UDIM + kbase + dt * 16 + nn] = (bf16)v;
        }
      }
      __syncthreads();  // protect S/SC before next head overwrites them
    }
  }

  // Write head-averaged scores once, with plain coalesced stores.
  float* avgRow = avg + ((size_t)n * TSEQ + q0 + row) * TSEQ;
  for (int t = sub; t < TSEQ; t += 16) avgRow[t] = AVG[row * TSEQ + t];
}

// ---------- launch ----------
extern "C" void kernel_launch(void* const* d_in, const int* in_sizes, int n_in,
                              void* d_out, int out_size, void* d_ws, size_t ws_size,
                              hipStream_t stream) {
  const float*   q_in = (const float*)d_in[0];
  const float*   k_in = (const float*)d_in[1];
  const float*   v_in = (const float*)d_in[2];
  const uint8_t* mask = (const uint8_t*)d_in[3];
  const float*   Wq   = (const float*)d_in[4];
  const float*   Wk   = (const float*)d_in[5];
  const float*   Wv   = (const float*)d_in[6];
  const float*   Wo   = (const float*)d_in[7];
  const float*   bo   = (const float*)d_in[8];

  float* outp = (float*)d_out;                       // [8192][1024] f32
  float* avgp = outp + (size_t)MROWS * UDIM;         // [4][2048][2048] f32

  char* ws = (char*)d_ws;
  size_t off = 0;
  auto alloc = [&](size_t bytes) -> void* {
    void* p = ws + off;
    off += (bytes + 255) & ~(size_t)255;
    return p;
  };
  const size_t big = (size_t)MROWS * UDIM * sizeof(bf16);  // 16 MB
  const size_t wsz = (size_t)UDIM * UDIM * sizeof(bf16);   // 2 MB
  bf16* qin = (bf16*)alloc(big);
  bf16* kin = (bf16*)alloc(big);
  bf16* vin = (bf16*)alloc(big);
  bf16* wqt = (bf16*)alloc(wsz);
  bf16* wkt = (bf16*)alloc(wsz);
  bf16* wvt = (bf16*)alloc(wsz);
  bf16* wot = (bf16*)alloc(wsz);
  bf16* Qp  = (bf16*)alloc(big);
  bf16* Kp  = (bf16*)alloc(big);
  bf16* Vt  = (bf16*)alloc(big);   // transposed: [1024 d][8192 t]
  bf16* ctx = qin;                 // qin dead after Q projection; reuse
  if (off > ws_size) return;       // workspace too small; bail

  const int nel = MROWS * UDIM;
  cast_f32_bf16<<<(nel + 255) / 256, 256, 0, stream>>>(q_in, qin, nel);
  cast_f32_bf16<<<(nel + 255) / 256, 256, 0, stream>>>(k_in, kin, nel);
  cast_f32_bf16<<<(nel + 255) / 256, 256, 0, stream>>>(v_in, vin, nel);
  const int welem = UDIM * UDIM;
  transpose_cast<<<(welem + 255) / 256, 256, 0, stream>>>(Wq, wqt, UDIM, UDIM);
  transpose_cast<<<(welem + 255) / 256, 256, 0, stream>>>(Wk, wkt, UDIM, UDIM);
  transpose_cast<<<(welem + 255) / 256, 256, 0, stream>>>(Wv, wvt, UDIM, UDIM);
  transpose_cast<<<(welem + 255) / 256, 256, 0, stream>>>(Wo, wot, UDIM, UDIM);

  dim3 ggrid(UDIM / 64, MROWS / 256);
  // Q projection with 1/sqrt(KEY_DIM)=1/32 folded in
  gemm_bf16_k<0><<<ggrid, 256, 0, stream>>>(qin, wqt, Qp, nullptr, UDIM, 0.03125f, UDIM);
  gemm_bf16_k<0><<<ggrid, 256, 0, stream>>>(kin, wkt, Kp, nullptr, UDIM, 1.0f, UDIM);
  gemm_bf16_k<1><<<ggrid, 256, 0, stream>>>(vin, wvt, Vt, nullptr, UDIM, 1.0f, MROWS);

  dim3 agrid(TSEQ / 16, NB);
  size_t ldsBytes = (size_t)(16 * TSEQ * 2 + 4 * 256) * sizeof(float);  // 260 KB
  attn_k<<<agrid, 256, ldsBytes, stream>>>(Qp, Kp, Vt, mask, avgp, ctx);

  gemm_bf16_k<2><<<ggrid, 256, 0, stream>>>(ctx, wot, outp, bo, UDIM, 1.0f, UDIM);
}
